// GCN_L_63745904607633
// MI455X (gfx1250) — matrix operand order repaired and checked by reading, not compiled, not run on hardware
//
#include <hip/hip_runtime.h>
#include <hip/hip_bf16.h>

// ---------------- problem constants (from reference) ----------------
#define N_     200000
#define D_     512
#define H_     512
#define B_     256
#define TOPK_  2
#define M_     100
#define NLAB_  2000
#define P_     (B_ * TOPK_)   // 512 pairs
#define ROWS_  (M_ + 1)       // x0 + members = 101

// ---------------- WMMA fragment types (gfx1250, wave32) ----------------
typedef _Float16 half8 __attribute__((ext_vector_type(8)));
typedef _Float16 v16h  __attribute__((ext_vector_type(16)));
typedef float    v8f   __attribute__((ext_vector_type(8)));

union H16 { v16h v; half8 h[2]; };

// A-matrix (16x32 f16, row-major source, stride K halfs):
//   lanes 0-15:  halfs 0..7 -> K = kb+0..7,  halfs 8..15 -> K = kb+16..23
//   lanes 16-31: halfs 0..7 -> K = kb+8..15, halfs 8..15 -> K = kb+24..31
__device__ __forceinline__ v16h load_fragA(const _Float16* __restrict__ A,
                                           int K, int row, int kb, int lane) {
  const int hi = (lane >= 16) ? 8 : 0;
  const _Float16* base = A + (long)row * K + kb + hi;
  H16 u;
  u.h[0] = *(const half8*)(base);
  u.h[1] = *(const half8*)(base + 16);
  return u.v;
}

// B-matrix (32x16 f16) sourced from an N-major (transposed) weight array,
// stride K halfs per output column:
//   lanes 0-15:  halfs 0..7 -> K = kb+0..7,  halfs 8..15 -> K = kb+16..23
//   lanes 16-31: halfs 0..7 -> K = kb+8..15, halfs 8..15 -> K = kb+24..31
__device__ __forceinline__ v16h load_fragB(const _Float16* __restrict__ Bt,
                                           int K, int col, int kb, int lane) {
  const int hi = (lane >= 16) ? 8 : 0;
  const _Float16* base = Bt + (long)col * K + kb + hi;
  H16 u;
  u.h[0] = *(const half8*)(base);
  u.h[1] = *(const half8*)(base + 16);
  return u.v;
}

// ---------------- kernel 0: transpose + f16-convert weights ----------------
__global__ __launch_bounds__(256) void k_convert(const float* __restrict__ W1,
                                                 const float* __restrict__ W2,
                                                 _Float16* __restrict__ W1t,
                                                 _Float16* __restrict__ W2t) {
  const int i = blockIdx.x * 256 + threadIdx.x;
  const int n1 = 2 * D_ * H_;                 // W1: (2D, H)
  if (i < n1) {
    const int d = i / H_, h = i % H_;
    W1t[(long)h * (2 * D_) + d] = (_Float16)W1[i];
  } else {
    const int j = i - n1;                     // W2: (H, H)
    if (j < H_ * H_) {
      const int h = j / H_, g = j % H_;
      W2t[(long)g * H_ + h] = (_Float16)W2[j];
    }
  }
}

// ---------------- kernel 1: per-pair gather + attention + cat0 ----------------
__global__ __launch_bounds__(256) void k_phase1(
    const float* __restrict__ features, const int* __restrict__ labels,
    const int* __restrict__ indexes,   const int* __restrict__ neighbors,
    const float* __restrict__ all_pred,
    const int* __restrict__ cluster_members,
    const unsigned char* __restrict__ cluster_mask,
    _Float16* __restrict__ Acat, int* __restrict__ cand_out,
    float* __restrict__ valid_out)
{
  __shared__ float x0s[D_];
  __shared__ float a0s[ROWS_ + 3];
  __shared__ int   rids[ROWS_ + 3];

  const int p = blockIdx.x;
  const int b = p / TOPK_;
  const int k = p % TOPK_;
  const int t = threadIdx.x;
  const int lane = t & 31;
  const int wave = t >> 5;                    // 8 waves

  const int nb   = neighbors[p];
  const int cand = labels[nb];
  const int idx0 = indexes[b];

  for (int d = t; d < D_; d += 256) x0s[d] = features[(long)idx0 * D_ + d];
  __syncthreads();

  // pass 1: a0[r] = mask(r) ? x0 . feat[r] : 0   (row 0 = x0 itself)
  for (int r = wave; r < ROWS_; r += 8) {
    int rid; bool msk;
    if (r == 0) { rid = idx0; msk = true; }
    else {
      const int mi = cluster_members[cand * M_ + (r - 1)];
      msk = (cluster_mask[cand * M_ + (r - 1)] != 0) && (mi != idx0);
      rid = mi;
    }
    const float* fr = features + (long)rid * D_;
    float s = 0.f;
    #pragma unroll
    for (int j = 0; j < D_ / 32; ++j) s += x0s[j * 32 + lane] * fr[j * 32 + lane];
    #pragma unroll
    for (int off = 16; off > 0; off >>= 1) s += __shfl_xor(s, off, 32);
    if (lane == 0) { a0s[r] = msk ? s : 0.f; rids[r] = rid; }
  }
  __syncthreads();

  // pass 2: agg[d] = sum_r a0[r] * feat[r][d]; emit cat0 = [x0, agg] in f16
  #pragma unroll
  for (int dd = 0; dd < D_ / 256; ++dd) {
    const int d = t + dd * 256;
    float agg = 0.f;
    for (int r = 0; r < ROWS_; ++r) {
      const float* fr = features + (long)rids[r] * D_;
      if (r + 1 < ROWS_)
        __builtin_prefetch(features + (long)rids[r + 1] * D_ + d, 0, 0);
      agg += a0s[r] * fr[d];
    }
    Acat[(long)p * (2 * D_) + d]      = (_Float16)x0s[d];
    Acat[(long)p * (2 * D_) + D_ + d] = (_Float16)agg;
  }

  if (t == 0) {
    cand_out[p] = cand;
    float v = 1.0f;
    if (k != 0) {
      const int  cand0 = labels[neighbors[b * TOPK_]];
      const bool sel1  = all_pred[p] >= all_pred[b * TOPK_];
      v = (sel1 && cand != cand0) ? 1.0f : 0.0f;
    }
    valid_out[p] = v;
  }
}

// ---------------- kernel 2: H1 = relu(cat0 @ W1 + b1)  [WMMA f16] ----------------
__global__ __launch_bounds__(128) void k_gemm1(const _Float16* __restrict__ A,
                                               const _Float16* __restrict__ W1t,
                                               const float* __restrict__ b1,
                                               _Float16* __restrict__ H1) {
  const int lane = threadIdx.x & 31;
  const int wv   = threadIdx.x >> 5;
  const int tile = blockIdx.x * 4 + wv;       // 1024 tiles of 16x16 over 512x512
  const int tm = tile >> 5;
  const int tn = tile & 31;
  const int rowA = tm * 16 + (lane & 15);
  const int colB = tn * 16 + (lane & 15);

  v8f c = {};
  for (int kb = 0; kb < 2 * D_; kb += 32) {
    const v16h a  = load_fragA(A,   2 * D_, rowA, kb, lane);
    const v16h bb = load_fragB(W1t, 2 * D_, colB, kb, lane);
    c = __builtin_amdgcn_wmma_f32_16x16x32_f16(false, a, false, bb,
                                               (short)0, c, false, false);
  }
  const float bias  = b1[colB];
  const int   rbase = tm * 16 + ((lane >= 16) ? 8 : 0);
  #pragma unroll
  for (int r = 0; r < 8; ++r) {
    float v = c[r] + bias;
    v = v > 0.f ? v : 0.f;
    H1[(long)(rbase + r) * H_ + colB] = (_Float16)v;
  }
}

// ---------------- kernel 3: Z = prelu(H1 @ W2 + b2)  [WMMA f16] ----------------
__global__ __launch_bounds__(128) void k_gemm2(const _Float16* __restrict__ H1,
                                               const _Float16* __restrict__ W2t,
                                               const float* __restrict__ b2,
                                               const float* __restrict__ prelu_a,
                                               float* __restrict__ Z) {
  const int lane = threadIdx.x & 31;
  const int wv   = threadIdx.x >> 5;
  const int tile = blockIdx.x * 4 + wv;
  const int tm = tile >> 5;
  const int tn = tile & 31;
  const int rowA = tm * 16 + (lane & 15);
  const int colB = tn * 16 + (lane & 15);

  v8f c = {};
  for (int kb = 0; kb < H_; kb += 32) {
    const v16h a  = load_fragA(H1,  H_, rowA, kb, lane);
    const v16h bb = load_fragB(W2t, H_, colB, kb, lane);
    c = __builtin_amdgcn_wmma_f32_16x16x32_f16(false, a, false, bb,
                                               (short)0, c, false, false);
  }
  const float bias  = b2[colB];
  const float alpha = prelu_a[colB];
  const int   rbase = tm * 16 + ((lane >= 16) ? 8 : 0);
  #pragma unroll
  for (int r = 0; r < 8; ++r) {
    float v = c[r] + bias;
    v = v >= 0.f ? v : alpha * v;
    Z[(long)(rbase + r) * H_ + colB] = v;
  }
}

// ---------------- kernel 4: logits / log-softmax / masked CE per pair ----------------
__global__ __launch_bounds__(256) void k_head(const float* __restrict__ Z,
                                              const float* __restrict__ W3,
                                              const float* __restrict__ b3,
                                              const int* __restrict__ cand,
                                              const float* __restrict__ valid,
                                              float* __restrict__ partials) {
  __shared__ float s0[256], s1[256];
  const int p = blockIdx.x, t = threadIdx.x;
  float l0 = 0.f, l1 = 0.f;
  for (int g = t; g < H_; g += 256) {
    const float z = Z[(long)p * H_ + g];
    l0 += z * W3[g * 2 + 0];
    l1 += z * W3[g * 2 + 1];
  }
  s0[t] = l0; s1[t] = l1;
  __syncthreads();
  for (int off = 128; off > 0; off >>= 1) {
    if (t < off) { s0[t] += s0[t + off]; s1[t] += s1[t + off]; }
    __syncthreads();
  }
  if (t == 0) {
    const float a  = s0[0] + b3[0];
    const float bb = s1[0] + b3[1];
    const int   gt = (cand[p] == -1) ? 1 : 0;
    const float mx = fmaxf(a, bb);
    const float lse = mx + logf(expf(a - mx) + expf(bb - mx));
    const float lp  = (gt == 0 ? a : bb) - lse;
    partials[p] = -lp * valid[p];
  }
}

// ---------------- kernel 5: deterministic final reduction ----------------
__global__ __launch_bounds__(256) void k_reduce(const float* __restrict__ partials,
                                                float* __restrict__ out) {
  __shared__ float s[256];
  const int t = threadIdx.x;
  float acc = 0.f;
  for (int i = t; i < P_; i += 256) acc += partials[i];
  s[t] = acc;
  __syncthreads();
  for (int off = 128; off > 0; off >>= 1) {
    if (t < off) s[t] += s[t + off];
    __syncthreads();
  }
  if (t == 0) out[0] = s[0] / (float)B_ * 10.0f;
}

// ---------------- launcher ----------------
extern "C" void kernel_launch(void* const* d_in, const int* in_sizes, int n_in,
                              void* d_out, int out_size, void* d_ws, size_t ws_size,
                              hipStream_t stream) {
  (void)in_sizes; (void)n_in; (void)out_size; (void)ws_size;

  const float* features        = (const float*)d_in[0];
  const int*   labels          = (const int*)d_in[1];
  const int*   indexes         = (const int*)d_in[2];
  const int*   neighbors       = (const int*)d_in[3];
  const float* all_pred        = (const float*)d_in[4];
  const int*   cluster_members = (const int*)d_in[5];
  const unsigned char* cluster_mask = (const unsigned char*)d_in[6];
  const float* W1 = (const float*)d_in[7];
  const float* b1 = (const float*)d_in[8];
  const float* W2 = (const float*)d_in[9];
  const float* b2 = (const float*)d_in[10];
  const float* prelu_a = (const float*)d_in[11];
  const float* W3 = (const float*)d_in[12];
  const float* b3 = (const float*)d_in[13];

  char* ws = (char*)d_ws;
  size_t off = 0;
  _Float16* Acat = (_Float16*)(ws + off); off += (size_t)P_ * 2 * D_ * sizeof(_Float16); // 1 MB
  _Float16* W1t  = (_Float16*)(ws + off); off += (size_t)H_ * 2 * D_ * sizeof(_Float16); // 1 MB
  _Float16* W2t  = (_Float16*)(ws + off); off += (size_t)H_ * H_ * sizeof(_Float16);     // 0.5 MB
  _Float16* H1   = (_Float16*)(ws + off); off += (size_t)P_ * H_ * sizeof(_Float16);     // 0.5 MB
  float*    Z    = (float*)(ws + off);    off += (size_t)P_ * H_ * sizeof(float);        // 1 MB
  int*      cand = (int*)(ws + off);      off += (size_t)P_ * sizeof(int);
  float*    valid= (float*)(ws + off);    off += (size_t)P_ * sizeof(float);
  float*    part = (float*)(ws + off);    off += (size_t)P_ * sizeof(float);

  // 0) weight convert/transpose to f16
  {
    const int total = 2 * D_ * H_ + H_ * H_;
    k_convert<<<(total + 255) / 256, 256, 0, stream>>>(W1, W2, W1t, W2t);
  }
  // 1) gather + attention + cat0
  k_phase1<<<P_, 256, 0, stream>>>(features, labels, indexes, neighbors, all_pred,
                                   cluster_members, cluster_mask, Acat, cand, valid);
  // 2) GEMM1 (512x1024 @ 1024x512) + relu  -> H1
  k_gemm1<<<256, 128, 0, stream>>>(Acat, W1t, b1, H1);
  // 3) GEMM2 (512x512 @ 512x512) + prelu   -> Z
  k_gemm2<<<256, 128, 0, stream>>>(H1, W2t, b2, prelu_a, Z);
  // 4) head + masked CE
  k_head<<<P_, 256, 0, stream>>>(Z, W3, b3, cand, valid, part);
  // 5) deterministic scalar loss
  k_reduce<<<1, 256, 0, stream>>>(part, (float*)d_out);
}
